// SNDShowerClassifier_20693152432521
// MI455X (gfx1250) — matrix operand order repaired and checked by reading, not compiled, not run on hardware
//
#include <hip/hip_runtime.h>
#include <hip/hip_bf16.h>

// ---------------------------------------------------------------------------
// SND shower classifier for gfx1250 (MI455X).
// Big GEMMs (N=131072 rows) run on v_wmma_f32_16x16x32_f16; irregular graph
// ops (KNN, segment stats, edge aggregation, softmax attention) are wave32
// VALU kernels.  f16 storage for GEMM operands, f32 accumulation throughout.
// ---------------------------------------------------------------------------

typedef __attribute__((ext_vector_type(16))) _Float16 v16h;
typedef __attribute__((ext_vector_type(8)))  _Float16 h8;
typedef __attribute__((ext_vector_type(8)))  float    v8f;

static constexpr int NB    = 256;     // events
static constexpr int NPE   = 512;     // hits per event
static constexpr int NTOT  = NB * NPE;          // 131072
static constexpr int KNN_K = 16;
static constexpr int HID   = 64;
static constexpr int PROP  = 32;
static constexpr int GSTAT = NB * 2 * 64;       // 32768 station groups
static constexpr int GMT_K = 5;
static constexpr int GMT_H = 4;
static constexpr int DH    = HID / GMT_H;       // 16

// ---------------- input pointer order (setup_inputs dict-insertion order) ---
enum {
  IN_X = 0, IN_ZQ, IN_XY, IN_ORI, IN_BIDX,
  G0_LS_W, G0_LS_B, G0_LH_W, G0_LH_B, G0_SKIP_W, G0_AGG_W, G0_AGG_B, G0_BN_G, G0_BN_B,
  G1_LS_W, G1_LS_B, G1_LH_W, G1_LH_B, G1_SKIP_W, G1_AGG_W, G1_AGG_B, G1_BN_G, G1_BN_B,
  GMT_S,
  PMA_Q_W, PMA_Q_B, PMA_K_W, PMA_K_B, PMA_V_W, PMA_V_B, PMA_O_W, PMA_O_B,
  SAB_Q_W, SAB_Q_B, SAB_K_W, SAB_K_B, SAB_V_W, SAB_V_B, SAB_O_W, SAB_O_B,
  MLP0_W, MLP0_B, MLP1_W, MLP1_B, MLP2_W, MLP2_B,
  IN_COUNT
};

// ---------------------------------------------------------------------------
__global__ void zero_kernel(float* p, int n) {
  int t = blockIdx.x * blockDim.x + threadIdx.x;
  if (t < n) p[t] = 0.0f;
}

// convert f32 weight [rows x cols] to f16 [rpad x cols] (zero-padded rows)
__global__ void cvt_w_kernel(const float* __restrict__ src, _Float16* __restrict__ dst,
                             int rows, int cols, int rpad) {
  int t = blockIdx.x * blockDim.x + threadIdx.x;
  if (t >= rpad * cols) return;
  int r = t / cols, c = t % cols;
  dst[t] = (r < rows) ? (_Float16)src[r * cols + c] : (_Float16)0.0f;
}

// ---------------- station features ----------------------------------------
// st layout: [0,G) cnt, [G,2G) sum(xy), [2G,3G) sum(xy^2)
__global__ void st_accum_kernel(const float* __restrict__ xy, const int* __restrict__ zq,
                                const int* __restrict__ ori, const int* __restrict__ bidx,
                                float* __restrict__ st) {
  int t = blockIdx.x * blockDim.x + threadIdx.x;
  if (t >= NTOT) return;
  int gid = bidx[t] * 128 + zq[t] * 2 + ori[t];
  float v = xy[t];
  atomicAdd(&st[gid], 1.0f);
  atomicAdd(&st[GSTAT + gid], v);
  atomicAdd(&st[2 * GSTAT + gid], v * v);
}

// h0 padded to [NTOT x 32] f16: cols 0..7 = x, 8=dist, 9=std, 10=frac, 11..31 = 0
__global__ void st_build_kernel(const float* __restrict__ x, const float* __restrict__ xy,
                                const int* __restrict__ zq, const int* __restrict__ ori,
                                const int* __restrict__ bidx, const float* __restrict__ st,
                                _Float16* __restrict__ h0) {
  int n = blockIdx.x * blockDim.x + threadIdx.x;
  if (n >= NTOT) return;
  int gid = bidx[n] * 128 + zq[n] * 2 + ori[n];
  float c    = st[gid];
  float dn   = fmaxf(c, 1.0f);
  float cent = st[GSTAT + gid] / dn;
  float sqm  = st[2 * GSTAT + gid] / dn;
  float sd   = sqrtf(fmaxf(sqm - cent * cent, 0.0f));
  float dist = fabsf(xy[n] - cent);
  float frac = c * (1.0f / (float)NPE);          // nev per event is exactly NPE
  _Float16* o = h0 + (size_t)n * 32;
  #pragma unroll
  for (int j = 0; j < 8; ++j) o[j] = (_Float16)x[(size_t)n * 8 + j];
  o[8] = (_Float16)dist; o[9] = (_Float16)sd; o[10] = (_Float16)frac;
  #pragma unroll
  for (int j = 11; j < 32; ++j) o[j] = (_Float16)0.0f;
}

// ---------------- small fused linears: s (4) and h (32) projections --------
__global__ void lin_sh_kernel(const _Float16* __restrict__ A, int lda, int ic,
                              const float* __restrict__ Ws, const float* __restrict__ bs,
                              const float* __restrict__ Wh, const float* __restrict__ bh,
                              float* __restrict__ S, float* __restrict__ HF) {
  int t = blockIdx.x * blockDim.x + threadIdx.x;
  if (t >= NTOT * 36) return;
  int n = t / 36, j = t % 36;
  const _Float16* ar = A + (size_t)n * lda;
  if (j < 4) {
    float acc = bs[j];
    for (int k = 0; k < ic; ++k) acc += (float)ar[k] * Ws[k * 4 + j];
    S[(size_t)n * 4 + j] = acc;
  } else {
    int jj = j - 4;
    float acc = bh[jj];
    for (int k = 0; k < ic; ++k) acc += (float)ar[k] * Wh[k * 32 + jj];
    HF[(size_t)n * 32 + jj] = acc;
  }
}

// ---------------- KNN (per-event 512x512 in 4-D learned space) -------------
__global__ void __launch_bounds__(512) knn_kernel(const float* __restrict__ s,
                                                  int* __restrict__ knn) {
  __shared__ float ss[NPE * 4];
  int b = blockIdx.x, i = threadIdx.x;
  const float* sb = s + (size_t)b * NPE * 4;
  #pragma unroll
  for (int d = 0; d < 4; ++d) ss[i * 4 + d] = sb[i * 4 + d];
  __syncthreads();
  float q0 = ss[i * 4 + 0], q1 = ss[i * 4 + 1], q2 = ss[i * 4 + 2], q3 = ss[i * 4 + 3];
  float bd[KNN_K]; int bi[KNN_K];
  #pragma unroll
  for (int k = 0; k < KNN_K; ++k) { bd[k] = 3.0e38f; bi[k] = 0; }
  for (int j = 0; j < NPE; ++j) {
    if (j == i) continue;                         // ref excludes self-loops
    float d0 = ss[j * 4 + 0] - q0, d1 = ss[j * 4 + 1] - q1;
    float d2 = ss[j * 4 + 2] - q2, d3 = ss[j * 4 + 3] - q3;
    float d = d0 * d0 + d1 * d1 + d2 * d2 + d3 * d3;
    if (d < bd[KNN_K - 1]) {
      bd[KNN_K - 1] = d; bi[KNN_K - 1] = j;
      #pragma unroll
      for (int t = KNN_K - 1; t > 0; --t) {
        if (bd[t] < bd[t - 1]) {
          float td = bd[t]; bd[t] = bd[t - 1]; bd[t - 1] = td;
          int   ti = bi[t]; bi[t] = bi[t - 1]; bi[t - 1] = ti;
        }
      }
    }
  }
  int base = (b * NPE + i) * KNN_K;
  #pragma unroll
  for (int k = 0; k < KNN_K; ++k) knn[base + k] = b * NPE + bi[k];
}

// ---------------- edge aggregation (each dst has exactly K=16 edges) -------
// agg16: [NTOT x 128] f16 = [mean | max | smooth | deriv] of 32 props each
__global__ void agg_kernel(const int* __restrict__ knn, const float* __restrict__ s,
                           const float* __restrict__ hf, const int* __restrict__ zq,
                           _Float16* __restrict__ agg16) {
  int t = blockIdx.x * blockDim.x + threadIdx.x;
  if (t >= NTOT * PROP) return;
  int n = t >> 5, p = t & 31;
  float s0 = s[(size_t)n * 4 + 0], s1 = s[(size_t)n * 4 + 1];
  float s2 = s[(size_t)n * 4 + 2], s3 = s[(size_t)n * 4 + 3];
  float zd = (float)zq[n];
  int srcv[KNN_K]; float ewv[KNN_K]; float dzv[KNN_K];
  float sumAbs = 0.0f;
  #pragma unroll
  for (int k = 0; k < KNN_K; ++k) {
    int si = knn[(size_t)n * KNN_K + k];
    srcv[k] = si;
    float d0 = s[(size_t)si * 4 + 0] - s0, d1 = s[(size_t)si * 4 + 1] - s1;
    float d2 = s[(size_t)si * 4 + 2] - s2, d3 = s[(size_t)si * 4 + 3] - s3;
    ewv[k] = __expf(-10.0f * (d0 * d0 + d1 * d1 + d2 * d2 + d3 * d3));
    float dz = (float)zq[si] - zd;
    dzv[k] = dz;
    sumAbs += fabsf(dz);
  }
  float inv = 1.0f / (sumAbs + 1e-8f);
  float sm = 0.0f, mx = -3.4e38f, ssm = 0.0f, sdv = 0.0f;
  #pragma unroll
  for (int k = 0; k < KNN_K; ++k) {
    float m = hf[(size_t)srcv[k] * PROP + p] * ewv[k];
    float F = dzv[k] * inv;
    sm += m; mx = fmaxf(mx, m); ssm += m * fabsf(F); sdv += m * F;
  }
  size_t o = (size_t)n * 128;
  agg16[o + p]      = (_Float16)(sm * (1.0f / 16.0f));
  agg16[o + 32 + p] = (_Float16)mx;
  agg16[o + 64 + p] = (_Float16)(ssm * (1.0f / 16.0f));
  agg16[o + 96 + p] = (_Float16)sdv;
}

// ---------------- WMMA GEMM: C[M x 64] (+)= A[M x KDIM] * W[KDIM x 64] -----
// A f16 row-major (lda = KDIM); W f16 row-major, transposed into LDS so each
// lane's B fragment (element e = W[kb + lh*16 + e][col]) is one contiguous
// 32-byte LDS segment -> 2x ds_load_b128.  All B fragments for the wave's
// N-tile are hoisted to VGPRs (<= 4 x v16h), then MT M-tiles are processed
// with zero LDS traffic in the steady state.
// Fragment k order is contiguous (k = lh*16 + e) in both operands (16-bit A 16x32; 16-bit B 32x16).
// OUTF16: 0 -> write f32 C only, 1 -> write f16 C16 only.
template <int KDIM, int MT, int OUTF16, int BETA, int HASB>
__global__ void __launch_bounds__(128) gemm_wmma_n64(
    const _Float16* __restrict__ A, const _Float16* __restrict__ W,
    const float* __restrict__ bias, float* __restrict__ C,
    _Float16* __restrict__ C16) {
  __shared__ _Float16 Wt[64 * KDIM];            // Wt[col][k] = W[k][col]
  for (int i = threadIdx.x; i < 64 * KDIM; i += 128) {
    int c = i / KDIM, k = i % KDIM;             // contiguous LDS stores
    Wt[i] = W[k * 64 + c];
  }
  __syncthreads();
  const int wave = threadIdx.x >> 5;
  const int lane = threadIdx.x & 31;
  const int lh   = lane >> 4;                   // lane half
  const int ln   = lane & 15;
  const int col  = wave * 16 + ln;              // 0..63
  constexpr int KSTEPS = KDIM / 32;

  // B fragments for all K-steps, kept in registers for the whole kernel.
  v16h bfrag[KSTEPS];
  #pragma unroll
  for (int s = 0; s < KSTEPS; ++s) {
    const h8* src = (const h8*)(Wt + col * KDIM + s * 32 + lh * 16);
    h8 blo = src[0], bhi = src[1];
    #pragma unroll
    for (int i = 0; i < 8; ++i) { bfrag[s][i] = blo[i]; bfrag[s][8 + i] = bhi[i]; }
  }
  float bc = 0.0f;
  if constexpr (HASB) bc = bias[col];

  #pragma unroll
  for (int mt = 0; mt < MT; ++mt) {
    const int m0 = (blockIdx.x * MT + mt) * 16;
    v8f c;
    if constexpr (BETA) {
      #pragma unroll
      for (int r = 0; r < 8; ++r) c[r] = C[(size_t)(m0 + r + 8 * lh) * 64 + col];
    } else {
      #pragma unroll
      for (int r = 0; r < 8; ++r) c[r] = 0.0f;
    }
    const _Float16* Ar = A + (size_t)(m0 + ln) * KDIM;
    #pragma unroll
    for (int s = 0; s < KSTEPS; ++s) {
      const int kb = s * 32;
      // A 16x32: lanes 0-15 K=0..15 ; lanes 16-31 K=16..31 (matches B)
      h8 lo = *(const h8*)(Ar + kb + lh * 16);
      h8 hi = *(const h8*)(Ar + kb + lh * 16 + 8);
      v16h a;
      #pragma unroll
      for (int i = 0; i < 8; ++i) { a[i] = lo[i]; a[8 + i] = hi[i]; }
      c = __builtin_amdgcn_wmma_f32_16x16x32_f16(false, a, false, bfrag[s],
                                                 (short)0, c, false, false);
    }
    #pragma unroll
    for (int r = 0; r < 8; ++r) {
      float v = c[r];
      if constexpr (HASB) v += bc;        // avoid dead +0.0 VALU when no bias
      size_t o = (size_t)(m0 + r + 8 * lh) * 64 + col;
      if constexpr (OUTF16) C16[o] = (_Float16)v;
      else                  C[o]   = v;
    }
  }
}

// ---------------- batch-norm over N (training-mode, biased var) ------------
__global__ void __launch_bounds__(256) bn_stats_kernel(const float* __restrict__ outf,
                                                       float* __restrict__ st) {
  __shared__ float ssum[64], ssq[64];
  int c = threadIdx.x & 63;
  if (threadIdx.x < 64) { ssum[threadIdx.x] = 0.0f; ssq[threadIdx.x] = 0.0f; }
  __syncthreads();
  float a = 0.0f, q = 0.0f;
  for (int row = blockIdx.x * 4 + (threadIdx.x >> 6); row < NTOT; row += gridDim.x * 4) {
    float v = outf[(size_t)row * 64 + c];
    a += v; q += v * v;
  }
  atomicAdd(&ssum[c], a); atomicAdd(&ssq[c], q);
  __syncthreads();
  if (threadIdx.x < 64) {
    atomicAdd(&st[threadIdx.x], ssum[threadIdx.x]);
    atomicAdd(&st[64 + threadIdx.x], ssq[threadIdx.x]);
  }
}

__global__ void bn_apply_kernel(const float* __restrict__ outf, const float* __restrict__ st,
                                const float* __restrict__ g, const float* __restrict__ b,
                                _Float16* __restrict__ h16) {
  int t = blockIdx.x * blockDim.x + threadIdx.x;
  if (t >= NTOT * 64) return;
  int c = t & 63;
  float mu  = st[c] * (1.0f / (float)NTOT);
  float var = st[64 + c] * (1.0f / (float)NTOT) - mu * mu;
  float v = (outf[t] - mu) * rsqrtf(var + 1e-5f) * g[c] + b[c];
  h16[t] = (_Float16)fmaxf(v, 0.0f);
}

// ---------------- generic small f32 linear (optionally relu / residual) ----
__global__ void lin_small_kernel(const float* __restrict__ in, const float* __restrict__ W,
                                 const float* __restrict__ bias, float* __restrict__ out,
                                 int R, int IC, int OC, int relu, int resid) {
  int t = blockIdx.x * blockDim.x + threadIdx.x;
  if (t >= R * OC) return;
  int r = t / OC, c = t % OC;
  float acc = bias ? bias[c] : 0.0f;
  const float* ir = in + (size_t)r * IC;
  for (int k = 0; k < IC; ++k) acc += ir[k] * W[k * OC + c];
  if (resid)      out[t] = in[(size_t)r * IC + c] + fmaxf(acc, 0.0f);  // x + relu(lin(x))
  else if (relu)  out[t] = fmaxf(acc, 0.0f);
  else            out[t] = acc;
}

// ---------------- PMA attention: 5 seeds attend to 512 hits (4 heads) ------
__global__ void pma_attn_kernel(const float* __restrict__ qf, const _Float16* __restrict__ k16,
                                const _Float16* __restrict__ v16, float* __restrict__ out1) {
  int t = blockIdx.x * blockDim.x + threadIdx.x;
  if (t >= NB * GMT_H * GMT_K) return;
  int qi = t % GMT_K;
  int h  = (t / GMT_K) % GMT_H;
  int b  = t / (GMT_K * GMT_H);
  float qv[DH];
  #pragma unroll
  for (int d = 0; d < DH; ++d) qv[d] = qf[qi * HID + h * DH + d];
  float mx = -3.4e38f, den = 0.0f, acc[DH];
  #pragma unroll
  for (int d = 0; d < DH; ++d) acc[d] = 0.0f;
  for (int m = 0; m < NPE; ++m) {
    const _Float16* kr = k16 + ((size_t)(b * NPE + m)) * HID + h * DH;
    float s = 0.0f;
    #pragma unroll
    for (int d = 0; d < DH; ++d) s += qv[d] * (float)kr[d];
    s *= 0.125f;                                   // / sqrt(HID)
    float nm = fmaxf(mx, s);
    float cs = __expf(mx - nm);
    float es = __expf(s - nm);
    const _Float16* vr = v16 + ((size_t)(b * NPE + m)) * HID + h * DH;
    den = den * cs + es;
    #pragma unroll
    for (int d = 0; d < DH; ++d) acc[d] = acc[d] * cs + es * (float)vr[d];
    mx = nm;
  }
  float invd = 1.0f / den;
  #pragma unroll
  for (int d = 0; d < DH; ++d)
    out1[((size_t)(b * GMT_K + qi)) * HID + h * DH + d] =
        qf[qi * HID + h * DH + d] + acc[d] * invd;
}

// ---------------- SAB attention over the 5 seeds ---------------------------
__global__ void sab_attn_kernel(const float* __restrict__ q2, const float* __restrict__ k2,
                                const float* __restrict__ v2, float* __restrict__ out2) {
  int t = blockIdx.x * blockDim.x + threadIdx.x;
  if (t >= NB * GMT_H * GMT_K) return;
  int qi = t % GMT_K;
  int h  = (t / GMT_K) % GMT_H;
  int b  = t / (GMT_K * GMT_H);
  const float* qr = q2 + ((size_t)(b * GMT_K + qi)) * HID + h * DH;
  float sc[GMT_K]; float mx = -3.4e38f;
  #pragma unroll
  for (int m = 0; m < GMT_K; ++m) {
    const float* kr = k2 + ((size_t)(b * GMT_K + m)) * HID + h * DH;
    float s = 0.0f;
    #pragma unroll
    for (int d = 0; d < DH; ++d) s += qr[d] * kr[d];
    s *= 0.125f;
    sc[m] = s; mx = fmaxf(mx, s);
  }
  float den = 0.0f;
  #pragma unroll
  for (int m = 0; m < GMT_K; ++m) { sc[m] = __expf(sc[m] - mx); den += sc[m]; }
  float invd = 1.0f / den;
  #pragma unroll
  for (int d = 0; d < DH; ++d) {
    float o = 0.0f;
    #pragma unroll
    for (int m = 0; m < GMT_K; ++m)
      o += sc[m] * v2[((size_t)(b * GMT_K + m)) * HID + h * DH + d];
    out2[((size_t)(b * GMT_K + qi)) * HID + h * DH + d] = qr[d] + o * invd;
  }
}

__global__ void pool_kernel(const float* __restrict__ p2, float* __restrict__ pooled) {
  int t = blockIdx.x * blockDim.x + threadIdx.x;
  if (t >= NB * HID) return;
  int b = t / HID, c = t % HID;
  float s = 0.0f;
  #pragma unroll
  for (int m = 0; m < GMT_K; ++m) s += p2[((size_t)(b * GMT_K + m)) * HID + c];
  pooled[t] = s * (1.0f / (float)GMT_K);
}

// ---------------------------------------------------------------------------
extern "C" void kernel_launch(void* const* d_in, const int* in_sizes, int n_in,
                              void* d_out, int out_size, void* d_ws, size_t ws_size,
                              hipStream_t stream) {
  (void)in_sizes; (void)n_in; (void)out_size; (void)ws_size;

  const float* x    = (const float*)d_in[IN_X];
  const int*   zq   = (const int*)d_in[IN_ZQ];
  const float* xy   = (const float*)d_in[IN_XY];
  const int*   ori  = (const int*)d_in[IN_ORI];
  const int*   bidx = (const int*)d_in[IN_BIDX];
  const float* Sf   = (const float*)d_in[GMT_S];

  // ---- workspace layout (assumes ws_size >= ~125 MB) ----
  char* wp = (char*)d_ws;
  auto alloc = [&](size_t bytes) -> void* {
    void* r = (void*)wp;
    wp += (bytes + 255) & ~(size_t)255;
    return r;
  };
  float*    stat   = (float*)alloc(3 * GSTAT * sizeof(float));
  float*    bnstat = (float*)alloc(128 * sizeof(float));
  _Float16* h0pad  = (_Float16*)alloc((size_t)NTOT * 32 * 2);
  _Float16* h16    = (_Float16*)alloc((size_t)NTOT * 64 * 2);
  float*    sbuf   = (float*)alloc((size_t)NTOT * 4 * 4);
  float*    hf     = (float*)alloc((size_t)NTOT * 32 * 4);
  int*      knn    = (int*)alloc((size_t)NTOT * KNN_K * 4);
  _Float16* agg16  = (_Float16*)alloc((size_t)NTOT * 128 * 2);
  float*    outf   = (float*)alloc((size_t)NTOT * 64 * 4);
  _Float16* w_sk0  = (_Float16*)alloc(32 * 64 * 2);
  _Float16* w_ag0  = (_Float16*)alloc(128 * 64 * 2);
  _Float16* w_sk1  = (_Float16*)alloc(64 * 64 * 2);
  _Float16* w_ag1  = (_Float16*)alloc(128 * 64 * 2);
  _Float16* w_k    = (_Float16*)alloc(64 * 64 * 2);
  _Float16* w_v    = (_Float16*)alloc(64 * 64 * 2);
  float*    qf     = (float*)alloc(GMT_K * HID * 4);
  float*    out1   = (float*)alloc((size_t)NB * GMT_K * HID * 4);
  float*    p1     = (float*)alloc((size_t)NB * GMT_K * HID * 4);
  float*    q2b    = (float*)alloc((size_t)NB * GMT_K * HID * 4);
  float*    k2b    = (float*)alloc((size_t)NB * GMT_K * HID * 4);
  float*    v2b    = (float*)alloc((size_t)NB * GMT_K * HID * 4);
  float*    out2   = (float*)alloc((size_t)NB * GMT_K * HID * 4);
  float*    p2     = (float*)alloc((size_t)NB * GMT_K * HID * 4);
  float*    pooled = (float*)alloc((size_t)NB * HID * 4);
  float*    f1     = (float*)alloc((size_t)NB * HID * 4);
  float*    f2     = (float*)alloc((size_t)NB * HID * 4);
  // k16/v16 alias agg16 (free once the layer-1 agg GEMM has consumed it)
  _Float16* k16 = agg16;
  _Float16* v16 = agg16 + (size_t)NTOT * 64;

  auto blocks = [](long n, int blk) { return (int)((n + blk - 1) / blk); };

  // ---- station features -> h0 ----
  zero_kernel<<<blocks(3 * GSTAT, 256), 256, 0, stream>>>(stat, 3 * GSTAT);
  st_accum_kernel<<<blocks(NTOT, 256), 256, 0, stream>>>(xy, zq, ori, bidx, stat);
  st_build_kernel<<<blocks(NTOT, 256), 256, 0, stream>>>(x, xy, zq, ori, bidx, stat, h0pad);

  // ---- weight conversion to f16 (padded) ----
  cvt_w_kernel<<<blocks(32 * 64, 256), 256, 0, stream>>>((const float*)d_in[G0_SKIP_W], w_sk0, 11, 64, 32);
  cvt_w_kernel<<<blocks(128 * 64, 256), 256, 0, stream>>>((const float*)d_in[G0_AGG_W], w_ag0, 128, 64, 128);
  cvt_w_kernel<<<blocks(64 * 64, 256), 256, 0, stream>>>((const float*)d_in[G1_SKIP_W], w_sk1, 64, 64, 64);
  cvt_w_kernel<<<blocks(128 * 64, 256), 256, 0, stream>>>((const float*)d_in[G1_AGG_W], w_ag1, 128, 64, 128);
  cvt_w_kernel<<<blocks(64 * 64, 256), 256, 0, stream>>>((const float*)d_in[PMA_K_W], w_k, 64, 64, 64);
  cvt_w_kernel<<<blocks(64 * 64, 256), 256, 0, stream>>>((const float*)d_in[PMA_V_W], w_v, 64, 64, 64);

  constexpr int MT = 4;                       // M-tiles per block
  const int gemm_grid = NTOT / (16 * MT);     // 2048 blocks, 128 threads (4 wave32)

  // ---- GravNet layer 0 (ic = 11, padded to 32) ----
  lin_sh_kernel<<<blocks((long)NTOT * 36, 256), 256, 0, stream>>>(
      h0pad, 32, 11,
      (const float*)d_in[G0_LS_W], (const float*)d_in[G0_LS_B],
      (const float*)d_in[G0_LH_W], (const float*)d_in[G0_LH_B], sbuf, hf);
  knn_kernel<<<NB, NPE, 0, stream>>>(sbuf, knn);
  agg_kernel<<<blocks((long)NTOT * PROP, 256), 256, 0, stream>>>(knn, sbuf, hf, zq, agg16);
  gemm_wmma_n64<32, MT, 0, 0, 0><<<gemm_grid, 128, 0, stream>>>(
      h0pad, w_sk0, nullptr, outf, nullptr);
  gemm_wmma_n64<128, MT, 0, 1, 1><<<gemm_grid, 128, 0, stream>>>(
      agg16, w_ag0, (const float*)d_in[G0_AGG_B], outf, nullptr);
  zero_kernel<<<1, 128, 0, stream>>>(bnstat, 128);
  bn_stats_kernel<<<2048, 256, 0, stream>>>(outf, bnstat);
  bn_apply_kernel<<<blocks((long)NTOT * 64, 256), 256, 0, stream>>>(
      outf, bnstat, (const float*)d_in[G0_BN_G], (const float*)d_in[G0_BN_B], h16);

  // ---- GravNet layer 1 (ic = 64) ----
  lin_sh_kernel<<<blocks((long)NTOT * 36, 256), 256, 0, stream>>>(
      h16, 64, 64,
      (const float*)d_in[G1_LS_W], (const float*)d_in[G1_LS_B],
      (const float*)d_in[G1_LH_W], (const float*)d_in[G1_LH_B], sbuf, hf);
  knn_kernel<<<NB, NPE, 0, stream>>>(sbuf, knn);
  agg_kernel<<<blocks((long)NTOT * PROP, 256), 256, 0, stream>>>(knn, sbuf, hf, zq, agg16);
  gemm_wmma_n64<64, MT, 0, 0, 0><<<gemm_grid, 128, 0, stream>>>(
      h16, w_sk1, nullptr, outf, nullptr);
  gemm_wmma_n64<128, MT, 0, 1, 1><<<gemm_grid, 128, 0, stream>>>(
      agg16, w_ag1, (const float*)d_in[G1_AGG_B], outf, nullptr);
  zero_kernel<<<1, 128, 0, stream>>>(bnstat, 128);
  bn_stats_kernel<<<2048, 256, 0, stream>>>(outf, bnstat);
  bn_apply_kernel<<<blocks((long)NTOT * 64, 256), 256, 0, stream>>>(
      outf, bnstat, (const float*)d_in[G1_BN_G], (const float*)d_in[G1_BN_B], h16);

  // ---- GMT: PMA (K/V over all hits via WMMA, Q from seeds) ----
  gemm_wmma_n64<64, MT, 1, 0, 1><<<gemm_grid, 128, 0, stream>>>(
      h16, w_k, (const float*)d_in[PMA_K_B], nullptr, k16);
  gemm_wmma_n64<64, MT, 1, 0, 1><<<gemm_grid, 128, 0, stream>>>(
      h16, w_v, (const float*)d_in[PMA_V_B], nullptr, v16);
  lin_small_kernel<<<blocks(GMT_K * HID, 64), 64, 0, stream>>>(
      Sf, (const float*)d_in[PMA_Q_W], (const float*)d_in[PMA_Q_B], qf,
      GMT_K, HID, HID, 0, 0);
  pma_attn_kernel<<<blocks(NB * GMT_H * GMT_K, 256), 256, 0, stream>>>(qf, k16, v16, out1);
  lin_small_kernel<<<blocks(NB * GMT_K * HID, 256), 256, 0, stream>>>(
      out1, (const float*)d_in[PMA_O_W], (const float*)d_in[PMA_O_B], p1,
      NB * GMT_K, HID, HID, 0, 1);

  // ---- GMT: SAB over the 5 seeds ----
  lin_small_kernel<<<blocks(NB * GMT_K * HID, 256), 256, 0, stream>>>(
      p1, (const float*)d_in[SAB_Q_W], (const float*)d_in[SAB_Q_B], q2b, NB * GMT_K, HID, HID, 0, 0);
  lin_small_kernel<<<blocks(NB * GMT_K * HID, 256), 256, 0, stream>>>(
      p1, (const float*)d_in[SAB_K_W], (const float*)d_in[SAB_K_B], k2b, NB * GMT_K, HID, HID, 0, 0);
  lin_small_kernel<<<blocks(NB * GMT_K * HID, 256), 256, 0, stream>>>(
      p1, (const float*)d_in[SAB_V_W], (const float*)d_in[SAB_V_B], v2b, NB * GMT_K, HID, HID, 0, 0);
  sab_attn_kernel<<<blocks(NB * GMT_H * GMT_K, 256), 256, 0, stream>>>(q2b, k2b, v2b, out2);
  lin_small_kernel<<<blocks(NB * GMT_K * HID, 256), 256, 0, stream>>>(
      out2, (const float*)d_in[SAB_O_W], (const float*)d_in[SAB_O_B], p2,
      NB * GMT_K, HID, HID, 0, 1);

  // ---- pool + MLP head ----
  pool_kernel<<<blocks(NB * HID, 256), 256, 0, stream>>>(p2, pooled);
  lin_small_kernel<<<blocks(NB * HID, 256), 256, 0, stream>>>(
      pooled, (const float*)d_in[MLP0_W], (const float*)d_in[MLP0_B], f1, NB, HID, HID, 1, 0);
  lin_small_kernel<<<blocks(NB * HID, 256), 256, 0, stream>>>(
      f1, (const float*)d_in[MLP1_W], (const float*)d_in[MLP1_B], f2, NB, HID, HID, 1, 0);
  lin_small_kernel<<<blocks(NB * 4, 256), 256, 0, stream>>>(
      f2, (const float*)d_in[MLP2_W], (const float*)d_in[MLP2_B], (float*)d_out, NB, HID, 4, 0, 0);
}
